// LSTMDecoder_66614942761493
// MI455X (gfx1250) — matrix operand order repaired and checked by reading, not compiled, not run on hardware
//
#include <hip/hip_runtime.h>
#include <hip/hip_bf16.h>

// ---------------------------------------------------------------------------
// LSTM caption decoder with additive attention, CDNA5 (gfx1250) bf16 WMMA.
// B=64 T=20 L=196 D=H=E=A=512 V=30000
// ---------------------------------------------------------------------------

#define Bn 64
#define Tn 20
#define Ln 196
#define Hn 512
#define Vn 30000

typedef __attribute__((ext_vector_type(8)))  float  v8f;
typedef __attribute__((ext_vector_type(8)))  __bf16 v8bf;
typedef __attribute__((ext_vector_type(16))) __bf16 v16bf;

// ---------------------------------------------------------------------------
// Generic bf16 WMMA GEMM:  C[M,N](f32, row stride ldc) = A[M,K] * BT[N,K]^T
//   A  : bf16 row-major [M,K]          (K contiguous -> A-fragment b128 loads)
//   BT : bf16 row-major [N,K] (== B col-major; K contiguous -> b256 B loads)
//   bias: optional [N]
// Each wave computes a 64x16 output strip; 8 waves / block.
// Requires: M%64==0, N%16==0, K%32==0.
// ---------------------------------------------------------------------------
__global__ void gemm_bf16_wmma(const __bf16* __restrict__ A,
                               const __bf16* __restrict__ BT,
                               const float* __restrict__ bias,
                               float* __restrict__ C,
                               int M, int N, int K, long long ldc)
{
    const int lane  = threadIdx.x & 31;
    const int wave  = threadIdx.x >> 5;
    const int ntile = blockIdx.x * 8 + wave;          // 16-col tile index
    if (ntile * 16 >= N) return;                      // wave-uniform exit
    const int m0   = blockIdx.y * 64;
    const int half = lane >> 4;                       // 0: lanes 0-15, 1: 16-31
    const int lid  = lane & 15;

    v8f acc[4] = {};                                  // 4 M-tiles of 16

    // B fragment: column n = ntile*16+lid, K contiguous (16 vals / k-step)
    const __bf16* Bp = BT + (size_t)(ntile * 16 + lid) * K + half * 16;
    // A fragment base: row m0+lid, contiguous K split (k = 16*half + s, same as B)
    const __bf16* Ap = A + (size_t)(m0 + lid) * K + half * 16;

    for (int k0 = 0; k0 < K; k0 += 32) {
        v16bf bfrag = *(const v16bf*)(Bp + k0);
#pragma unroll
        for (int mt = 0; mt < 4; ++mt) {
            const __bf16* ap = Ap + (size_t)(mt * 16) * K + k0;
            v8bf alo = *(const v8bf*)(ap);            // K: base+0..7  (or 16..23)
            v8bf ahi = *(const v8bf*)(ap + 8);        // K: base+8..15 (or 24..31)
            v16bf afrag = __builtin_shufflevector(alo, ahi,
                0,1,2,3,4,5,6,7,8,9,10,11,12,13,14,15);
            acc[mt] = __builtin_amdgcn_wmma_f32_16x16x32_bf16(
                false, afrag, false, bfrag, (short)0, acc[mt], false, false);
        }
    }

    const int n = ntile * 16 + lid;
    const float bv = bias ? bias[n] : 0.0f;
#pragma unroll
    for (int mt = 0; mt < 4; ++mt) {
        const int mbase = m0 + mt * 16 + half * 8;    // C/D VGPR->row mapping
#pragma unroll
        for (int j = 0; j < 8; ++j)
            C[(size_t)(mbase + j) * ldc + n] = acc[mt][j] + bv;
    }
}

// ---------------------------------------------------------------------------
// One-time weight preparation kernels
// ---------------------------------------------------------------------------

// Wcat[2048][1536] = [W_ih(2048x1024) | W_hh(2048x512)] as bf16 (used as B^T)
__global__ void build_wcat(const float* __restrict__ W_ih,
                           const float* __restrict__ W_hh,
                           __bf16* __restrict__ Wcat)
{
    int idx = blockIdx.x * 256 + threadIdx.x;         // 2048*1536
    int n = idx / 1536, k = idx - n * 1536;
    float v = (k < 1024) ? W_ih[(size_t)n * 1024 + k]
                         : W_hh[(size_t)n * 512 + (k - 1024)];
    Wcat[idx] = (__bf16)v;
}

// WqcT[512][1024]: WqcT[a][k] = k<512 ? Wq[k][a]+Wm[k][a] : Wc[k-512][a]
__global__ void build_wqct(const float* __restrict__ Wq,
                           const float* __restrict__ Wm,
                           const float* __restrict__ Wc,
                           __bf16* __restrict__ WqcT)
{
    int idx = blockIdx.x * 256 + threadIdx.x;         // 512*1024
    int a = idx >> 10, k = idx & 1023;
    float v = (k < 512) ? (Wq[(size_t)k * 512 + a] + Wm[(size_t)k * 512 + a])
                        : Wc[(size_t)(k - 512) * 512 + a];
    WqcT[idx] = (__bf16)v;
}

// Tiled transpose + f32->bf16:  dst[c*R + r] = src[r*C + c]
__global__ void transpose_cvt(const float* __restrict__ src,
                              __bf16* __restrict__ dst, int R, int C)
{
    __shared__ float tile[32][33];
    const int c0 = blockIdx.x * 32, r0 = blockIdx.y * 32;
    const int tx = threadIdx.x, ty = threadIdx.y;     // block (32,8)
#pragma unroll
    for (int i = 0; i < 32; i += 8) {
        int r = r0 + ty + i, c = c0 + tx;
        tile[ty + i][tx] = (r < R && c < C) ? src[(size_t)r * C + c] : 0.f;
    }
    __syncthreads();
#pragma unroll
    for (int i = 0; i < 32; i += 8) {
        int c = c0 + ty + i, r = r0 + tx;
        if (c < C && r < R) dst[(size_t)c * R + r] = (__bf16)tile[tx][ty + i];
    }
}

__global__ void cvt_bf16(const float* __restrict__ src,
                         __bf16* __restrict__ dst, int n)
{
    int i = blockIdx.x * 256 + threadIdx.x;
    if (i < n) dst[i] = (__bf16)src[i];
}

__global__ void zero_f32(float* __restrict__ p, int n)
{
    int i = blockIdx.x * 256 + threadIdx.x;
    if (i < n) p[i] = 0.f;
}

// ---------------------------------------------------------------------------
// Per-timestep glue kernels
// ---------------------------------------------------------------------------

// xh[b][0:512]=embedding[captions[b,t]], [512:1024]=ctx, [1024:1536]=h  (bf16)
__global__ void pack_x(const int* __restrict__ captions,
                       const float* __restrict__ embedding,
                       const float* __restrict__ ctx,
                       const float* __restrict__ h,
                       __bf16* __restrict__ xh, int t)
{
    int idx = blockIdx.x * 256 + threadIdx.x;         // 64*1536
    int b = idx / 1536, k = idx - b * 1536;
    float v;
    if (k < 512)       v = embedding[(size_t)captions[b * Tn + t] * 512 + k];
    else if (k < 1024) v = ctx[b * 512 + (k - 512)];
    else               v = h[b * 512 + (k - 1024)];
    xh[idx] = (__bf16)v;
}

__device__ __forceinline__ float sigf(float x) { return 1.f / (1.f + __expf(-x)); }

// gates[64,2048] (i|f|g|o) -> h,c (fp32, in place) + hc_bf16[64,1024]=[h;c]
__global__ void lstm_pointwise(const float* __restrict__ gates,
                               float* __restrict__ h, float* __restrict__ c,
                               __bf16* __restrict__ hc)
{
    int idx = blockIdx.x * 256 + threadIdx.x;         // 64*512
    int b = idx >> 9, a = idx & 511;
    const float* g = gates + (size_t)b * 2048;
    float iv = sigf(g[a]);
    float fv = sigf(g[512 + a]);
    float gv = tanhf(g[1024 + a]);
    float ov = sigf(g[1536 + a]);
    float cn = fv * c[idx] + iv * gv;
    float hn = ov * tanhf(cn);
    c[idx] = cn;
    h[idx] = hn;
    hc[(size_t)b * 1024 + a]       = (__bf16)hn;
    hc[(size_t)b * 1024 + 512 + a] = (__bf16)cn;
}

// Additive attention + softmax + context. One block per batch element.
__global__ void attention_kernel(const float* __restrict__ key_proj, // [B,L,A]
                                 const float* __restrict__ q,        // [B,A]
                                 const float* __restrict__ img,      // [B,L,D]
                                 const float* __restrict__ v_att,    // [A]
                                 const unsigned char* __restrict__ mask, // [B,L]
                                 float* __restrict__ ctx_f32,        // [B,D]
                                 __bf16* __restrict__ ctx_bf16)      // [B,D]
{
    __shared__ float s_w[256];
    const int b = blockIdx.x;
    const int tid = threadIdx.x;
    const int lane = tid & 31, wave = tid >> 5;

    // scores: one wave per image location, lanes reduce over A
    for (int l = wave; l < Ln; l += 8) {
        const float* kp = key_proj + ((size_t)b * Ln + l) * 512;
        float p = 0.f;
        for (int a = lane; a < 512; a += 32)
            p += tanhf(kp[a] + q[b * 512 + a]) * v_att[a];
        for (int off = 16; off > 0; off >>= 1)
            p += __shfl_down(p, off, 32);
        if (lane == 0)
            s_w[l] = mask[b * Ln + l] ? p : -3.4e38f;
    }
    __syncthreads();

    // softmax over L=196 (wave 0)
    if (wave == 0) {
        float mx = -3.4e38f;
        for (int l = lane; l < Ln; l += 32) mx = fmaxf(mx, s_w[l]);
        for (int off = 16; off > 0; off >>= 1)
            mx = fmaxf(mx, __shfl_xor(mx, off, 32));
        float sum = 0.f;
        for (int l = lane; l < Ln; l += 32) {
            float e = __expf(s_w[l] - mx);
            s_w[l] = e;
            sum += e;
        }
        for (int off = 16; off > 0; off >>= 1)
            sum += __shfl_xor(sum, off, 32);
        float inv = 1.f / sum;
        for (int l = lane; l < Ln; l += 32) s_w[l] *= inv;
    }
    __syncthreads();

    // context: ctx[d] = sum_l w[l] * img[b,l,d]   (coalesced over d)
    for (int d = tid; d < 512; d += 256) {
        float acc = 0.f;
        const float* fp = img + (size_t)b * Ln * 512 + d;
        for (int l = 0; l < Ln; ++l)
            acc += s_w[l] * fp[(size_t)l * 512];
        ctx_f32[b * 512 + d]  = acc;
        ctx_bf16[b * 512 + d] = (__bf16)acc;
    }
}

// ---------------------------------------------------------------------------
// Host-side orchestration
// ---------------------------------------------------------------------------
extern "C" void kernel_launch(void* const* d_in, const int* in_sizes, int n_in,
                              void* d_out, int out_size, void* d_ws, size_t ws_size,
                              hipStream_t stream)
{
    const int*   captions  = (const int*)  d_in[0];
    const float* img       = (const float*)d_in[1];   // [B,L,D]
    const float* pooled    = (const float*)d_in[2];   // [B,H]
    const unsigned char* mask = (const unsigned char*)d_in[3]; // [B,L] bool
    const float* embedding = (const float*)d_in[4];   // [V,E]
    const float* W_ih      = (const float*)d_in[5];
    const float* W_hh      = (const float*)d_in[6];
    const float* b_lstm    = (const float*)d_in[7];
    const float* Wq        = (const float*)d_in[8];
    const float* Wk        = (const float*)d_in[9];
    const float* Wm        = (const float*)d_in[10];
    const float* Wc        = (const float*)d_in[11];
    const float* v_att     = (const float*)d_in[12];
    const float* W_out     = (const float*)d_in[13];  // [H,V]
    const float* b_out     = (const float*)d_in[14];
    const float* W_init_h  = (const float*)d_in[15];
    const float* b_init_h  = (const float*)d_in[16];
    const float* W_init_c  = (const float*)d_in[17];
    const float* b_init_c  = (const float*)d_in[18];
    float* out = (float*)d_out;                       // [B,T,V]

    // ---- workspace carve (256B aligned) ----
    char* p = (char*)d_ws;
    auto alloc = [&](size_t bytes) -> void* {
        void* r = (void*)p;
        p += (bytes + 255) & ~(size_t)255;
        return r;
    };
    __bf16* Wcat     = (__bf16*)alloc((size_t)2048 * 1536 * 2); // [4H][E+2H]
    __bf16* WqcT     = (__bf16*)alloc((size_t)512 * 1024 * 2);  // [A][2H]
    __bf16* WoutT    = (__bf16*)alloc((size_t)Vn * 512 * 2);    // [V][H]
    __bf16* WkT      = (__bf16*)alloc((size_t)512 * 512 * 2);   // [A][D]
    __bf16* WihT     = (__bf16*)alloc((size_t)512 * 512 * 2);   // init_h^T
    __bf16* WicT     = (__bf16*)alloc((size_t)512 * 512 * 2);   // init_c^T
    __bf16* pooled_b = (__bf16*)alloc((size_t)Bn * 512 * 2);
    __bf16* img_b    = (__bf16*)alloc((size_t)Bn * Ln * 512 * 2);
    float*  key_proj = (float*) alloc((size_t)Bn * Ln * 512 * 4);
    __bf16* xh       = (__bf16*)alloc((size_t)Bn * 1536 * 2);
    float*  gates    = (float*) alloc((size_t)Bn * 2048 * 4);
    float*  h_f32    = (float*) alloc((size_t)Bn * 512 * 4);
    float*  c_f32    = (float*) alloc((size_t)Bn * 512 * 4);
    __bf16* hc_b     = (__bf16*)alloc((size_t)Bn * 1024 * 2);
    float*  q_f32    = (float*) alloc((size_t)Bn * 512 * 4);
    float*  ctx_f32  = (float*) alloc((size_t)Bn * 512 * 4);
    __bf16* ctx_b    = (__bf16*)alloc((size_t)Bn * 512 * 2);

    // ---- one-time weight prep ----
    build_wcat<<<12288, 256, 0, stream>>>(W_ih, W_hh, Wcat);
    build_wqct<<<2048, 256, 0, stream>>>(Wq, Wm, Wc, WqcT);
    transpose_cvt<<<dim3((Vn + 31) / 32, 16), dim3(32, 8), 0, stream>>>(W_out, WoutT, 512, Vn);
    transpose_cvt<<<dim3(16, 16), dim3(32, 8), 0, stream>>>(Wk, WkT, 512, 512);
    transpose_cvt<<<dim3(16, 16), dim3(32, 8), 0, stream>>>(W_init_h, WihT, 512, 512);
    transpose_cvt<<<dim3(16, 16), dim3(32, 8), 0, stream>>>(W_init_c, WicT, 512, 512);
    cvt_bf16<<<(Bn * 512 + 255) / 256, 256, 0, stream>>>(pooled, pooled_b, Bn * 512);
    cvt_bf16<<<(Bn * Ln * 512 + 255) / 256, 256, 0, stream>>>(img, img_b, Bn * Ln * 512);
    zero_f32<<<(Bn * 512 + 255) / 256, 256, 0, stream>>>(ctx_f32, Bn * 512);

    // ---- initial states: h0/c0 = pooled @ W_init + b ----
    gemm_bf16_wmma<<<dim3(4, 1), 256, 0, stream>>>(pooled_b, WihT, b_init_h,
                                                   h_f32, Bn, 512, 512, 512);
    gemm_bf16_wmma<<<dim3(4, 1), 256, 0, stream>>>(pooled_b, WicT, b_init_c,
                                                   c_f32, Bn, 512, 512, 512);
    // ---- key projection (hoisted): [B*L,512] @ WkT^T ----
    gemm_bf16_wmma<<<dim3(4, Bn * Ln / 64), 256, 0, stream>>>(img_b, WkT, nullptr,
                                                   key_proj, Bn * Ln, 512, 512, 512);

    // ---- timestep loop (sequential dependence; kernels serialize on stream)
    for (int t = 0; t < Tn; ++t) {
        pack_x<<<Bn * 1536 / 256, 256, 0, stream>>>(captions, embedding,
                                                    ctx_f32, h_f32, xh, t);
        // gates = [emb;ctx;h] @ [W_ih|W_hh]^T + b_lstm
        gemm_bf16_wmma<<<dim3(16, 1), 256, 0, stream>>>(xh, Wcat, b_lstm,
                                                        gates, Bn, 2048, 1536, 2048);
        lstm_pointwise<<<Bn * 512 / 256, 256, 0, stream>>>(gates, h_f32, c_f32, hc_b);
        // q = [h;c] @ [Wq+Wm ; Wc]
        gemm_bf16_wmma<<<dim3(4, 1), 256, 0, stream>>>(hc_b, WqcT, nullptr,
                                                       q_f32, Bn, 512, 1024, 512);
        attention_kernel<<<Bn, 256, 0, stream>>>(key_proj, q_f32, img, v_att,
                                                 mask, ctx_f32, ctx_b);
        // logits_t = ctx @ W_out + b_out  -> out[:, t, :]
        gemm_bf16_wmma<<<dim3((Vn / 16 + 7) / 8, 1), 256, 0, stream>>>(
            ctx_b, WoutT, b_out, out + (size_t)t * Vn,
            Bn, Vn, 512, (long long)Tn * Vn);
    }
    (void)in_sizes; (void)n_in; (void)out_size; (void)ws_size;
}
